// GraphSem_no_enc_agent_58334245814483
// MI455X (gfx1250) — matrix operand order repaired and checked by reading, not compiled, not run on hardware
//
#include <hip/hip_runtime.h>
#include <hip/hip_bf16.h>

// ---------------------------------------------------------------------------
// Problem constants
// ---------------------------------------------------------------------------
#define CB   4096          // batch
#define CN   32            // agents
#define CH   128           // hidden
#define CIN  96            // input feat
#define CA   12            // actions
#define CL   2             // decoder layers
#define CFF  256           // ffn dim
#define CBN  (CB * CN)     // 131072 rows

// output layout (floats): q | h_out | uniform_weights
#define QOFF  0
#define HOFF  ((size_t)CBN * CA)                   // 1,572,864
#define UOFF  (HOFF + (size_t)CBN * CH)            // 18,350,080

// f16 workspace offsets (in _Float16 elements, all 16B aligned)
#define O_FC1  0
#define O_WIH  12288
#define O_WHH  61440
#define O_FC2  110592
#define O_WV   634880
#define O_WO   667648
#define O_F1   700416
#define O_F2   765952
#define O_W1   831488
#define O_W2   864256
#define O_H16  866304                      // h_out f16 copy (131072*128)
#define O_D16  (O_H16 + (size_t)CBN * CH)  // dec f16 (4096*128)
#define H_TOT  (O_D16 + (size_t)CB * CH)
#define HBYTES ((size_t)H_TOT * 2)
// f32 region after the halves: enc (4096*128)
#define F_ENC  0

typedef __attribute__((ext_vector_type(16))) _Float16 v16h;
typedef __attribute__((ext_vector_type(8)))  _Float16 v8h;
typedef __attribute__((ext_vector_type(8)))  float    v8f;
typedef __attribute__((ext_vector_type(4)))  unsigned int u32x4;
typedef __attribute__((ext_vector_type(8)))  int      i32x8;
typedef __attribute__((ext_vector_type(4)))  int      i32x4;

// ---------------------------------------------------------------------------
// WMMA helpers (v_wmma_f32_16x16x32_f16, wave32, 16x16 tile per wave)
// ---------------------------------------------------------------------------
__device__ __forceinline__ v8f wmma_f16(v16h a, v16h b, v8f c) {
  return __builtin_amdgcn_wmma_f32_16x16x32_f16(
      false, a, false, b, (short)0, c, false, false);
}

// A fragment (16xK tile, row-major f16 in LDS, ld multiple of 8, k0 mult of 32)
__device__ __forceinline__ v16h frag_a_lds(const _Float16* Ab, int ld, int k0) {
  int lane = threadIdx.x & 31;
  int m = lane & 15, hf = lane >> 4;
  const _Float16* p = Ab + m * ld + k0 + hf * 8;
  v8h lo = *(const v8h*)p;
  v8h hi = *(const v8h*)(p + 16);
  v16h a;
#pragma unroll
  for (int i = 0; i < 8; ++i) { a[i] = lo[i]; a[i + 8] = hi[i]; }
  return a;
}

// B fragment from global f16 weights stored (out,in) row-major == [N][K]
__device__ __forceinline__ v16h frag_b_glb(const _Float16* __restrict__ W,
                                           int ld, int n0, int k0) {
  int lane = threadIdx.x & 31;
  int n = lane & 15, hf = lane >> 4;
  const _Float16* p = W + (size_t)(n0 + n) * ld + k0 + hf * 8;
  v8h lo = *(const v8h*)p;
  v8h hi = *(const v8h*)(p + 16);
  v16h b;
#pragma unroll
  for (int i = 0; i < 8; ++i) { b[i] = lo[i]; b[i + 8] = hi[i]; }
  return b;
}

// ---------------------------------------------------------------------------
// CDNA5 async / tensor data movement
// ---------------------------------------------------------------------------
// per-lane 16B global -> LDS async copy (ASYNCcnt)
__device__ __forceinline__ void async_cp16(void* lds_dst, const void* gsrc) {
  unsigned l = (unsigned)(size_t)lds_dst;     // flat low 32 bits == LDS address
  asm volatile("global_load_async_to_lds_b128 %0, %1, off"
               :: "v"(l), "v"(gsrc) : "memory");
}
__device__ __forceinline__ void wait_async0() {
  asm volatile("s_wait_asynccnt 0x0" ::: "memory");
}

// TDM: load a 16x128 f16 tile (row stride 4096 halves) from global into LDS.
// D# per CDNA5 ISA 8.3/8.4: group0 = {count, lds_addr, global_addr, type=2},
// group1 = {data_size=2B, tensor dims/strides, tile dims}.
__device__ __forceinline__ void tdm_load_tile16x128(_Float16* lds_dst,
                                                    const _Float16* gsrc) {
  unsigned lds = (unsigned)(size_t)lds_dst;
  unsigned long long ga = (unsigned long long)(size_t)gsrc;
  u32x4 g0;
  g0[0] = 1u;                                           // count=1 (valid D#)
  g0[1] = lds;                                          // lds_addr
  g0[2] = (unsigned)ga;                                 // global_addr[31:0]
  g0[3] = (unsigned)((ga >> 32) & 0x1FFFFFFu) | (2u << 30);  // addr[56:32]|type=2
  i32x8 g1;
  g1[0] = 0x10000;                         // workgroup_mask=0, data_size=1 (2B)
  g1[1] = (4096 & 0xFFFF) << 16;           // tensor_dim0[15:0] @ bits 63:48
  g1[2] = (4096 >> 16) | ((4096 & 0xFFFF) << 16);  // dim0 hi | tensor_dim1 lo
  g1[3] = (4096 >> 16) | (128 << 16);      // dim1 hi | tile_dim0 = 128
  g1[4] = 16;                              // tile_dim1 = 16, tile_dim2 = 0
  g1[5] = 4096;                            // tensor_dim0_stride[31:0]
  g1[6] = 0;                               // stride0 hi | stride1 lo
  g1[7] = 0;                               // stride1 hi
  i32x4 z = {};
#if defined(__clang_major__) && (__clang_major__ >= 23)
  i32x8 z8 = {};
  __builtin_amdgcn_tensor_load_to_lds(g0, g1, z, z, z8, 0);
#else
  __builtin_amdgcn_tensor_load_to_lds(g0, g1, z, z, 0);
#endif
}

__device__ __forceinline__ float fsigmoid(float x) {
  return 1.f / (1.f + __expf(-x));
}
__device__ __forceinline__ float ftanh_fast(float x) {
  float u = fminf(fmaxf(x, -15.f), 15.f);
  float t = __expf(2.f * u);
  return (t - 1.f) / (t + 1.f);
}

// ---------------------------------------------------------------------------
// weight conversion prologue
// ---------------------------------------------------------------------------
__global__ void cvt_f32_f16(const float* __restrict__ src,
                            _Float16* __restrict__ dst, int n, float scale) {
  int i = blockIdx.x * 256 + threadIdx.x;
  if (i < n) dst[i] = (_Float16)(src[i] * scale);
}

__global__ void cvt_pad_w2(const float* __restrict__ src,
                           _Float16* __restrict__ dst) {
  int i = blockIdx.x * 256 + threadIdx.x;
  if (i < 16 * 128)
    dst[i] = (i < CA * CH) ? (_Float16)src[i] : (_Float16)0.f;
}

// ---------------------------------------------------------------------------
// Kernel 1: fc1 + ReLU + GRU cell -> h_out (f32) + h16 (f16 copy)
// ---------------------------------------------------------------------------
__global__ __launch_bounds__(256) void k1_fc1_gru(
    const float* __restrict__ inputs, const float* __restrict__ hidden,
    const _Float16* __restrict__ fc1_wh, const float* __restrict__ fc1_b,
    const _Float16* __restrict__ wih_h, const _Float16* __restrict__ whh_h,
    const float* __restrict__ bih, const float* __restrict__ bhh,
    float* __restrict__ hout, _Float16* __restrict__ h16) {
  __shared__ _Float16 Ain[16 * CIN];   // inputs tile  f16
  __shared__ _Float16 Hh[16 * CH];     // h_in tile    f16
  __shared__ float    Hf[16 * CH];     // h_in tile    f32 (async-copied)
  __shared__ _Float16 Xs[16 * CH];     // relu(fc1)    f16

  const int tid = threadIdx.x;
  const int r0 = blockIdx.x * 16;

  // async raw-copy h_in tile (8KB = 512 x 16B), overlapped with Ain convert
  for (int c = tid; c < 512; c += 256) {
    int row = c >> 5, off = (c & 31) * 4;
    async_cp16(&Hf[row * CH + off], hidden + (size_t)(r0 + row) * CH + off);
  }
  for (int i = tid; i < 16 * CIN; i += 256) {
    int m = i / CIN, k = i % CIN;
    Ain[i] = (_Float16)inputs[(size_t)(r0 + m) * CIN + k];
  }
  wait_async0();
  __syncthreads();
  for (int i = tid; i < 16 * CH; i += 256) Hh[i] = (_Float16)Hf[i];
  __syncthreads();

  const int w = tid >> 5, lane = tid & 31;
  const int n = lane & 15, mb = (lane >> 4) * 8;
  const int col = w * 16 + n;

  // fc1 + ReLU
  v8f xa = {};
#pragma unroll
  for (int k0 = 0; k0 < CIN; k0 += 32)
    xa = wmma_f16(frag_a_lds(Ain, CIN, k0), frag_b_glb(fc1_wh, CIN, w * 16, k0), xa);
  {
    float bb = fc1_b[col];
#pragma unroll
    for (int r = 0; r < 8; ++r) {
      float v = xa[r] + bb; v = v > 0.f ? v : 0.f;
      Xs[(mb + r) * CH + col] = (_Float16)v;
    }
  }
  __syncthreads();

  // gi = x @ Wih^T, gh = h @ Whh^T  (3 gate blocks each)
  v8f gi[3], gh[3];
#pragma unroll
  for (int g = 0; g < 3; ++g) { gi[g] = v8f{}; gh[g] = v8f{}; }
#pragma unroll
  for (int g = 0; g < 3; ++g) {
    int nc = g * CH + w * 16;
#pragma unroll
    for (int k0 = 0; k0 < CH; k0 += 32) {
      gi[g] = wmma_f16(frag_a_lds(Xs, CH, k0), frag_b_glb(wih_h, CH, nc, k0), gi[g]);
      gh[g] = wmma_f16(frag_a_lds(Hh, CH, k0), frag_b_glb(whh_h, CH, nc, k0), gh[g]);
    }
  }
  const float bi0 = bih[col], bi1 = bih[CH + col], bi2 = bih[2 * CH + col];
  const float bh0 = bhh[col], bh1 = bhh[CH + col], bh2 = bhh[2 * CH + col];
#pragma unroll
  for (int r = 0; r < 8; ++r) {
    float ir = gi[0][r] + bi0, iz = gi[1][r] + bi1, in = gi[2][r] + bi2;
    float hr = gh[0][r] + bh0, hz = gh[1][r] + bh1, hn = gh[2][r] + bh2;
    float rr = fsigmoid(ir + hr);
    float zz = fsigmoid(iz + hz);
    float nn = ftanh_fast(in + rr * hn);
    float hp = Hf[(mb + r) * CH + col];
    float ho = (1.f - zz) * nn + zz * hp;
    size_t oi = (size_t)(r0 + mb + r) * CH + col;
    hout[oi] = ho;
    h16[oi] = (_Float16)ho;
  }
}

// ---------------------------------------------------------------------------
// Kernel 2: enc = relu((h/N) @ fc2_w^T + b)  K=4096, TDM double-buffered A
// ---------------------------------------------------------------------------
__global__ __launch_bounds__(256) void k2_fc2(
    const _Float16* __restrict__ h16, const _Float16* __restrict__ fc2_wh,
    const float* __restrict__ fc2_b, float* __restrict__ enc) {
  __shared__ _Float16 As[2][16 * 128];
  const int tid = threadIdx.x;
  const int b0 = blockIdx.x * 16;
  const int w = tid >> 5, lane = tid & 31;
  const int n = lane & 15, mb = (lane >> 4) * 8;
  const _Float16* Arow = h16 + (size_t)b0 * (CN * CH);

  if (w == 0) tdm_load_tile16x128(&As[0][0], Arow);

  v8f acc = {};
  for (int c = 0; c < 32; ++c) {
    if (w == 0) __builtin_amdgcn_s_wait_tensorcnt(0);  // oldest DMA done
    __syncthreads();                                   // tile visible; reads of
    if (w == 0 && c + 1 < 32)                          // other buf finished
      tdm_load_tile16x128(&As[(c + 1) & 1][0], Arow + (c + 1) * 128);
    const _Float16* Ab = &As[c & 1][0];
#pragma unroll
    for (int k0 = 0; k0 < 128; k0 += 32)
      acc = wmma_f16(frag_a_lds(Ab, 128, k0),
                     frag_b_glb(fc2_wh, CN * CH, w * 16, c * 128 + k0), acc);
  }
  float bb = fc2_b[w * 16 + n];
#pragma unroll
  for (int r = 0; r < 8; ++r) {
    float v = acc[r] + bb; v = v > 0.f ? v : 0.f;
    enc[(size_t)(b0 + mb + r) * CH + w * 16 + n] = v;
  }
}

// ---------------------------------------------------------------------------
// Kernel 3: decoder on enc (4096 rows; attention collapses to V->O path)
// ---------------------------------------------------------------------------
__device__ __forceinline__ void ln_tile(const float* Ys, float* Xs, _Float16* Xh,
                                        float* rs, float* rq, float* mu, float* ri,
                                        const float* __restrict__ g,
                                        const float* __restrict__ bta, int tid) {
  int row = tid >> 4, l16 = tid & 15;
  const float* p = Ys + row * CH + l16 * 8;
  float s = 0.f, q = 0.f;
#pragma unroll
  for (int i = 0; i < 8; ++i) { float v = p[i]; s += v; q += v * v; }
  rs[tid] = s; rq[tid] = q;
  __syncthreads();
  if (l16 == 0) {
    float S = 0.f, Q = 0.f;
#pragma unroll
    for (int i = 0; i < 16; ++i) { S += rs[row * 16 + i]; Q += rq[row * 16 + i]; }
    float m = S * (1.f / CH);
    float var = Q * (1.f / CH) - m * m;
    mu[row] = m; ri[row] = rsqrtf(var + 1e-5f);
  }
  __syncthreads();
  float m = mu[row], r = ri[row];
#pragma unroll
  for (int i = 0; i < 8; ++i) {
    int c = l16 * 8 + i;
    float v = (Ys[row * CH + c] - m) * r * g[c] + bta[c];
    Xs[row * CH + c] = v; Xh[row * CH + c] = (_Float16)v;
  }
  __syncthreads();
}

__global__ __launch_bounds__(256) void k3_decoder(
    const float* __restrict__ enc,
    const _Float16* __restrict__ wv_h, const float* __restrict__ bv,
    const _Float16* __restrict__ wo_h, const float* __restrict__ bo,
    const float* __restrict__ ln1_g, const float* __restrict__ ln1_b,
    const _Float16* __restrict__ f1_h, const float* __restrict__ fb1,
    const _Float16* __restrict__ f2_h, const float* __restrict__ fb2,
    const float* __restrict__ ln2_g, const float* __restrict__ ln2_b,
    _Float16* __restrict__ dec16) {
  __shared__ float    Xs[16 * CH];
  __shared__ float    Ys[16 * CH];
  __shared__ _Float16 Xh[16 * CH];
  __shared__ _Float16 Th[16 * CFF];
  __shared__ float rs[256], rq[256], mu[16], ri[16];

  const int tid = threadIdx.x;
  const int b0 = blockIdx.x * 16;
  const int w = tid >> 5, lane = tid & 31;
  const int n = lane & 15, mb = (lane >> 4) * 8;
  const int col = w * 16 + n;

  for (int i = tid; i < 16 * CH; i += 256) {
    float v = enc[(size_t)b0 * CH + i];
    Xs[i] = v; Xh[i] = (_Float16)v;
  }
  __syncthreads();

  for (int l = 0; l < CL; ++l) {
    const _Float16* Wv = wv_h + l * CH * CH;
    const _Float16* Wo = wo_h + l * CH * CH;
    const _Float16* W1 = f1_h + l * CFF * CH;
    const _Float16* W2 = f2_h + l * CH * CFF;

    // V = X @ Wv^T + bv   (uniform attention => ctx == V)
    v8f acc = {};
#pragma unroll
    for (int k0 = 0; k0 < CH; k0 += 32)
      acc = wmma_f16(frag_a_lds(Xh, CH, k0), frag_b_glb(Wv, CH, w * 16, k0), acc);
    {
      float bb = bv[l * CH + col];
#pragma unroll
      for (int r = 0; r < 8; ++r)
        Th[(mb + r) * CH + col] = (_Float16)(acc[r] + bb);
    }
    __syncthreads();

    // Y = X + V @ Wo^T + bo
    acc = v8f{};
#pragma unroll
    for (int k0 = 0; k0 < CH; k0 += 32)
      acc = wmma_f16(frag_a_lds(Th, CH, k0), frag_b_glb(Wo, CH, w * 16, k0), acc);
    {
      float bb = bo[l * CH + col];
#pragma unroll
      for (int r = 0; r < 8; ++r)
        Ys[(mb + r) * CH + col] = Xs[(mb + r) * CH + col] + acc[r] + bb;
    }
    __syncthreads();
    ln_tile(Ys, Xs, Xh, rs, rq, mu, ri, ln1_g + l * CH, ln1_b + l * CH, tid);

    // F1 = relu(X @ W1^T + b1)   (256 outputs: 2 tiles per wave)
    v8f f0 = {}, f1 = {};
#pragma unroll
    for (int k0 = 0; k0 < CH; k0 += 32) {
      v16h a = frag_a_lds(Xh, CH, k0);
      f0 = wmma_f16(a, frag_b_glb(W1, CH, w * 16, k0), f0);
      f1 = wmma_f16(a, frag_b_glb(W1, CH, CH + w * 16, k0), f1);
    }
    {
      float bb0 = fb1[l * CFF + col], bb1 = fb1[l * CFF + CH + col];
#pragma unroll
      for (int r = 0; r < 8; ++r) {
        float v0 = f0[r] + bb0; v0 = v0 > 0.f ? v0 : 0.f;
        float v1 = f1[r] + bb1; v1 = v1 > 0.f ? v1 : 0.f;
        Th[(mb + r) * CFF + col] = (_Float16)v0;
        Th[(mb + r) * CFF + CH + col] = (_Float16)v1;
      }
    }
    __syncthreads();

    // Y = X + F1 @ W2^T + b2
    acc = v8f{};
#pragma unroll
    for (int k0 = 0; k0 < CFF; k0 += 32)
      acc = wmma_f16(frag_a_lds(Th, CFF, k0), frag_b_glb(W2, CFF, w * 16, k0), acc);
    {
      float bb = fb2[l * CH + col];
#pragma unroll
      for (int r = 0; r < 8; ++r)
        Ys[(mb + r) * CH + col] = Xs[(mb + r) * CH + col] + acc[r] + bb;
    }
    __syncthreads();
    ln_tile(Ys, Xs, Xh, rs, rq, mu, ri, ln2_g + l * CH, ln2_b + l * CH, tid);
  }

  for (int i = tid; i < 16 * CH; i += 256)
    dec16[(size_t)b0 * CH + i] = (_Float16)Xs[i];
}

// ---------------------------------------------------------------------------
// Kernel 4: fc3 (concat(h_out, dec) -> relu -> q) + uniform_weights fill
// A tiles are pure f16 copies -> async global->LDS, zero staging VALU
// ---------------------------------------------------------------------------
__global__ __launch_bounds__(256) void k4_fc3(
    const _Float16* __restrict__ h16, const _Float16* __restrict__ dec16,
    const _Float16* __restrict__ w1h, const float* __restrict__ b1,
    const _Float16* __restrict__ w2h, const float* __restrict__ b2,
    float* __restrict__ q, float* __restrict__ uw) {
  __shared__ _Float16 Hh[16 * CH];
  __shared__ _Float16 Dh[16 * CH];
  __shared__ _Float16 Th[16 * CH];
  const int tid = threadIdx.x;
  const int r0 = blockIdx.x * 16;
  const int bidx = r0 >> 5;                 // constant per 16-row tile
  const int w = tid >> 5, lane = tid & 31;
  const int n = lane & 15, mb = (lane >> 4) * 8;

  {
    int row = tid >> 4, off = (tid & 15) * 8;     // 16B per thread
    async_cp16(&Hh[row * CH + off], h16 + (size_t)(r0 + row) * CH + off);
    async_cp16(&Dh[row * CH + off], dec16 + (size_t)bidx * CH + off);  // bcast
  }
  wait_async0();
  __syncthreads();

  // t1 = relu([h|dec] @ W1^T + b1), W1 is (128, 256): cols [0:128)=h, rest=dec
  v8f acc = {};
#pragma unroll
  for (int k0 = 0; k0 < CH; k0 += 32) {
    acc = wmma_f16(frag_a_lds(Hh, CH, k0), frag_b_glb(w1h, 2 * CH, w * 16, k0), acc);
    acc = wmma_f16(frag_a_lds(Dh, CH, k0), frag_b_glb(w1h, 2 * CH, w * 16, CH + k0), acc);
  }
  {
    float bb = b1[w * 16 + n];
#pragma unroll
    for (int r = 0; r < 8; ++r) {
      float v = acc[r] + bb; v = v > 0.f ? v : 0.f;
      Th[(mb + r) * CH + w * 16 + n] = (_Float16)v;
    }
  }
  __syncthreads();

  if (w == 0) {  // wave 0 (full EXEC) computes the 16x16 q tile (12 valid cols)
    v8f qa = {};
#pragma unroll
    for (int k0 = 0; k0 < CH; k0 += 32)
      qa = wmma_f16(frag_a_lds(Th, CH, k0), frag_b_glb(w2h, CH, 0, k0), qa);
    if (n < CA) {
      float bb = b2[n];
#pragma unroll
      for (int r = 0; r < 8; ++r)
        q[(size_t)(r0 + mb + r) * CA + n] = qa[r] + bb;
    }
  }

  // uniform_weights: 8192 blocks * 256 threads * 2 floats == 4,194,304 exactly
  size_t base = ((size_t)blockIdx.x * 256 + tid) * 2;
  float2 v2; v2.x = 1.f / CN; v2.y = 1.f / CN;
  *(float2*)(uw + base) = v2;
}

// ---------------------------------------------------------------------------
// launch
// ---------------------------------------------------------------------------
extern "C" void kernel_launch(void* const* d_in, const int* in_sizes, int n_in,
                              void* d_out, int out_size, void* d_ws, size_t ws_size,
                              hipStream_t stream) {
  const float* inputs = (const float*)d_in[0];
  const float* hidden = (const float*)d_in[1];
  const float* fc1_w  = (const float*)d_in[2];
  const float* fc1_b  = (const float*)d_in[3];
  const float* w_ih   = (const float*)d_in[4];
  const float* w_hh   = (const float*)d_in[5];
  const float* b_ih   = (const float*)d_in[6];
  const float* b_hh   = (const float*)d_in[7];
  const float* fc2_w  = (const float*)d_in[8];
  const float* fc2_b  = (const float*)d_in[9];
  // d_in[10..13] = wq, bq, wk, bk : provably unused (uniform attention)
  const float* wv     = (const float*)d_in[14];
  const float* bv     = (const float*)d_in[15];
  const float* wo     = (const float*)d_in[16];
  const float* bo     = (const float*)d_in[17];
  const float* ln1_g  = (const float*)d_in[18];
  const float* ln1_b  = (const float*)d_in[19];
  const float* ffn_w1 = (const float*)d_in[20];
  const float* ffn_b1 = (const float*)d_in[21];
  const float* ffn_w2 = (const float*)d_in[22];
  const float* ffn_b2 = (const float*)d_in[23];
  const float* ln2_g  = (const float*)d_in[24];
  const float* ln2_b  = (const float*)d_in[25];
  const float* fc3_w1 = (const float*)d_in[26];
  const float* fc3_b1 = (const float*)d_in[27];
  const float* fc3_w2 = (const float*)d_in[28];
  const float* fc3_b2 = (const float*)d_in[29];

  float* out  = (float*)d_out;
  float* qout = out + QOFF;
  float* hout = out + HOFF;
  float* uwt  = out + UOFF;

  _Float16* Wh = (_Float16*)d_ws;
  float* enc = (float*)((char*)d_ws + HBYTES);
  _Float16* h16p = Wh + O_H16;
  _Float16* d16p = Wh + O_D16;

  auto cvt = [&](const float* s, _Float16* d, int nel, float sc) {
    cvt_f32_f16<<<(nel + 255) / 256, 256, 0, stream>>>(s, d, nel, sc);
  };
  cvt(fc1_w,  Wh + O_FC1, CH * CIN, 1.f);
  cvt(w_ih,   Wh + O_WIH, 3 * CH * CH, 1.f);
  cvt(w_hh,   Wh + O_WHH, 3 * CH * CH, 1.f);
  cvt(fc2_w,  Wh + O_FC2, CH * CN * CH, 1.f / CN);   // fold 1/N comm scale
  cvt(wv,     Wh + O_WV,  CL * CH * CH, 1.f);
  cvt(wo,     Wh + O_WO,  CL * CH * CH, 1.f);
  cvt(ffn_w1, Wh + O_F1,  CL * CFF * CH, 1.f);
  cvt(ffn_w2, Wh + O_F2,  CL * CH * CFF, 1.f);
  cvt(fc3_w1, Wh + O_W1,  CH * 2 * CH, 1.f);
  cvt_pad_w2<<<(16 * CH + 255) / 256, 256, 0, stream>>>(fc3_w2, Wh + O_W2);

  k1_fc1_gru<<<CBN / 16, 256, 0, stream>>>(
      inputs, hidden, Wh + O_FC1, fc1_b, Wh + O_WIH, Wh + O_WHH,
      b_ih, b_hh, hout, h16p);

  k2_fc2<<<CB / 16, 256, 0, stream>>>(h16p, Wh + O_FC2, fc2_b, enc);

  k3_decoder<<<CB / 16, 256, 0, stream>>>(
      enc, Wh + O_WV, bv, Wh + O_WO, bo, ln1_g, ln1_b,
      Wh + O_F1, ffn_b1, Wh + O_F2, ffn_b2, ln2_g, ln2_b, d16p);

  k4_fc3<<<CBN / 16, 256, 0, stream>>>(
      h16p, d16p, Wh + O_W1, fc3_b1, Wh + O_W2, fc3_b2, qout, uwt);
}
